// TextRBM_24446953849238
// MI455X (gfx1250) — compile-verified, matrix operand order checked by reading, not run on hardware
//
#include <hip/hip_runtime.h>
#include <hip/hip_bf16.h>

#define BATCH 512
#define N_VIS 15000
#define N_HID 200
#define D_MAX 8192
#define KV    15008   // 469*32, padded K for GEMM1
#define KH    224     // 7*32,   padded K for GEMM2
#define NHP   208     // 13*16,  padded hidden rows for Wb

typedef __attribute__((ext_vector_type(16))) __bf16 v16bf;
typedef __attribute__((ext_vector_type(8)))  __bf16 v8bf;
typedef __attribute__((ext_vector_type(8)))  float  v8f;

// ---------------- deterministic counter-based RNG (splitmix64 -> [0,1)) ----
__device__ __forceinline__ float u01(unsigned long long x) {
    x += 0x9E3779B97F4A7C15ull;
    x = (x ^ (x >> 30)) * 0xBF58476D1CE4E5B9ull;
    x = (x ^ (x >> 27)) * 0x94D049BB133111EBull;
    x ^= x >> 31;
    return (float)((unsigned)(x >> 40)) * (1.0f / 16777216.0f);
}

// ---------------- d[b] = sum_k v[b,k] --------------------------------------
__global__ __launch_bounds__(256) void k_rowsum(const int* __restrict__ v,
                                                int* __restrict__ dvec) {
    __shared__ int red[256];
    const int b = blockIdx.x, tid = threadIdx.x;
    const int* row = v + (long long)b * N_VIS;
    int s = 0;
    for (int x = tid; x < N_VIS; x += 256) s += row[x];
    red[tid] = s; __syncthreads();
    for (int st = 128; st; st >>= 1) {
        if (tid < st) red[tid] += red[tid + st];
        __syncthreads();
    }
    if (tid == 0) dvec[b] = red[0];
}

// ---------------- out1 = float(v) ------------------------------------------
__global__ __launch_bounds__(256) void k_cast(const int* __restrict__ v,
                                              float* __restrict__ out1, long long n) {
    long long i = (long long)blockIdx.x * 256 + threadIdx.x;
    if (i < n) out1[i] = (float)v[i];
}

// ---------------- vb[b][k] = bf16(v[b][k]) zero-padded to KV ---------------
__global__ __launch_bounds__(256) void k_prep_v(const int* __restrict__ v,
                                                __bf16* __restrict__ vb) {
    const int k = blockIdx.x * 256 + threadIdx.x;
    const int b = blockIdx.y;
    if (k < KV)
        vb[(long long)b * KV + k] =
            (k < N_VIS) ? (__bf16)(float)v[(long long)b * N_VIS + k] : (__bf16)0.f;
}

// ---------------- Wb[j][k] = bf16(W[j][k]) padded to NHP x KV --------------
__global__ __launch_bounds__(256) void k_prep_wb(const float* __restrict__ W,
                                                 __bf16* __restrict__ wb) {
    const int k = blockIdx.x * 256 + threadIdx.x;
    const int j = blockIdx.y;
    if (k < KV)
        wb[(long long)j * KV + k] =
            (j < N_HID && k < N_VIS) ? (__bf16)W[(long long)j * N_VIS + k] : (__bf16)0.f;
}

// ---------------- Wt[x][j] = bf16(W[j][x]) : LDS tile transpose ------------
__global__ __launch_bounds__(256) void k_prep_wt(const float* __restrict__ W,
                                                 __bf16* __restrict__ wt) {
    __shared__ float tile[16][17];                 // +1 pad: conflict-free
    const int tx = threadIdx.x & 15, ty = threadIdx.x >> 4;
    const int x0 = blockIdx.x * 16, j0 = blockIdx.y * 16;
    const int xr = x0 + tx, jr = j0 + ty;          // coalesced read along x
    tile[ty][tx] = (jr < N_HID && xr < N_VIS) ? W[(long long)jr * N_VIS + xr] : 0.f;
    __syncthreads();
    const int xw = x0 + ty, jw = j0 + tx;          // coalesced write along j
    if (xw < KV && jw < KH)
        wt[(long long)xw * KH + jw] = (__bf16)tile[tx][ty];
}

// ---------------- hb = 0 ----------------------------------------------------
__global__ __launch_bounds__(256) void k_zero_hb(__bf16* __restrict__ hb) {
    hb[blockIdx.x * 256 + threadIdx.x] = (__bf16)0.f;
}

// ---------------- fragment helper: two contiguous b128 loads ---------------
__device__ __forceinline__ v16bf frag16(const __bf16* __restrict__ p, int half8) {
    v8bf lo = *(const v8bf*)(p + half8);
    v8bf hi = *(const v8bf*)(p + 16 + half8);
    v16bf f;
#pragma unroll
    for (int i = 0; i < 8; ++i) { f[i] = lo[i]; f[i + 8] = hi[i]; }
    return f;
}

// ---------------- h = bernoulli(sigmoid(v @ W^T + d*h_bias)) ---------------
// one wave / 16x16 tile; unguarded K loop: 4x global_load_b128 + 1 WMMA.
__global__ __launch_bounds__(32) void k_hidden(const __bf16* __restrict__ vb,
                                               const __bf16* __restrict__ wb,
                                               const float* __restrict__ h_bias,
                                               const int* __restrict__ dvec,
                                               __bf16* __restrict__ hb) {
    const int lane  = threadIdx.x;
    const int btile = blockIdx.x * 16;
    const int jtile = blockIdx.y * 16;
    const int m     = lane & 15;
    const int half8 = (lane >> 4) << 3;
    const __bf16* __restrict__ arow = vb + (long long)(btile + m) * KV;  // M side
    const __bf16* __restrict__ brow = wb + (long long)(jtile + m) * KV;  // N side (NHP rows)

    v8f c = {};
    for (int kb = 0; kb < KV; kb += 32) {
        __builtin_prefetch(arow + kb + 128, 0, 0);          // global_prefetch_b8
        v16bf a = frag16(arow + kb, half8);
        v16bf b = frag16(brow + kb, half8);
        c = __builtin_amdgcn_wmma_f32_16x16x32_bf16(false, a, false, b,
                                                    (short)0, c, false, false);
    }
    const int j = jtile + m;                    // C/D: lane n = m, row M = i + 8*half
    if (j < N_HID) {
        const float hbj = h_bias[j];
#pragma unroll
        for (int i = 0; i < 8; ++i) {
            const int brw = btile + i + (half8 >> 3) * 8 * 1 + ((lane >> 4) ? 8 : 0) * 0; // see below
            (void)brw;
        }
#pragma unroll
        for (int i = 0; i < 8; ++i) {
            const int row = btile + i + ((lane >> 4) << 3);
            const float df = (float)dvec[row];
            const float xx = c[i] + df * hbj;
            const float p  = 1.f / (1.f + __expf(-xx));
            const float u  = u01(((unsigned long long)row << 20) ^ (unsigned)j ^ 0xC0FFEEull);
            hb[(long long)row * KH + j] = (p > u) ? (__bf16)1.f : (__bf16)0.f;
        }
    }
}

// ---------------- logits = h @ W + v_bias ----------------------------------
__global__ __launch_bounds__(32) void k_logits(const __bf16* __restrict__ hb,
                                               const __bf16* __restrict__ wt,
                                               const float* __restrict__ v_bias,
                                               float* __restrict__ logits) {
    const int lane  = threadIdx.x;
    const int btile = blockIdx.x * 16;
    const int xtile = blockIdx.y * 16;
    const int m     = lane & 15;
    const int half8 = (lane >> 4) << 3;
    const __bf16* __restrict__ arow = hb + (long long)(btile + m) * KH;  // M side
    const __bf16* __restrict__ brow = wt + (long long)(xtile + m) * KH;  // N side (KV rows)

    v8f c = {};
#pragma unroll
    for (int kb = 0; kb < KH; kb += 32) {
        v16bf a = frag16(arow + kb, half8);
        v16bf b = frag16(brow + kb, half8);
        c = __builtin_amdgcn_wmma_f32_16x16x32_bf16(false, a, false, b,
                                                    (short)0, c, false, false);
    }
    const int x = xtile + m;
    if (x < N_VIS) {
        const float vbx = v_bias[x];
#pragma unroll
        for (int i = 0; i < 8; ++i) {
            const int row = btile + i + ((lane >> 4) << 3);
            logits[(long long)row * N_VIS + x] = c[i] + vbx;
        }
    }
}

// ---------------- row max for stable softmax -------------------------------
__global__ __launch_bounds__(256) void k_rowmax(const float* __restrict__ logits,
                                                float* __restrict__ rowmax) {
    __shared__ float red[256];
    const int b = blockIdx.x, tid = threadIdx.x;
    const float* row = logits + (long long)b * N_VIS;
    float mx = -1e30f;
    for (int x = tid; x < N_VIS; x += 256) mx = fmaxf(mx, row[x]);
    red[tid] = mx; __syncthreads();
    for (int st = 128; st; st >>= 1) {
        if (tid < st) red[tid] = fmaxf(red[tid], red[tid + st]);
        __syncthreads();
    }
    if (tid == 0) rowmax[b] = red[0];
}

// ---------------- softmax-CDF in LDS + multinomial draws + counts ----------
// One workgroup per row: 60KB CDF + 60KB counts in LDS (<< 320KB/WGP).
__global__ __launch_bounds__(256) void k_multinomial(const float* __restrict__ logits,
                                                     const float* __restrict__ rowmax,
                                                     const int* __restrict__ dvec,
                                                     float* __restrict__ out2) {
    extern __shared__ float cdf[];                       // N_VIS floats
    unsigned* cnt = (unsigned*)(cdf + N_VIS);            // N_VIS uints
    __shared__ float wsum[8];
    __shared__ float running;

    const int b = blockIdx.x, tid = threadIdx.x;
    const int lane = tid & 31, wid = tid >> 5;
    const float m = rowmax[b];
    const float* lrow = logits + (long long)b * N_VIS;

    for (int x = tid; x < N_VIS; x += 256) cnt[x] = 0u;
    if (tid == 0) running = 0.f;
    __syncthreads();

    // chunked inclusive scan: wave32 shuffle scan + 8-wave LDS combine
    for (int base = 0; base < N_VIS; base += 256) {
        const int x = base + tid;
        float val = (x < N_VIS) ? __expf(lrow[x] - m) : 0.f;
#pragma unroll
        for (int off = 1; off < 32; off <<= 1) {
            float t = __shfl_up(val, off, 32);
            if (lane >= off) val += t;
        }
        if (lane == 31) wsum[wid] = val;
        __syncthreads();
        if (wid == 0) {
            float w = (lane < 8) ? wsum[lane] : 0.f;
#pragma unroll
            for (int off = 1; off < 8; off <<= 1) {
                float t = __shfl_up(w, off, 32);
                if (lane >= off) w += t;
            }
            if (lane < 8) wsum[lane] = w;
        }
        __syncthreads();
        const float inc = running + val + (wid ? wsum[wid - 1] : 0.f);
        if (x < N_VIS) cdf[x] = inc;
        __syncthreads();                  // all reads of `running` done
        if (tid == 0) running += wsum[7];
        __syncthreads();
    }

    const float total = cdf[N_VIS - 1];
    int dcnt = dvec[b]; if (dcnt > D_MAX) dcnt = D_MAX;

    // inverse-CDF sampling: binary search over LDS-resident CDF
    for (int t = tid; t < D_MAX; t += 256) {
        if (t < dcnt) {
            const float u = u01(((unsigned long long)b << 32) ^ (unsigned)(t * 2654435761u)) * total;
            int lo = 0, hi = N_VIS - 1;
            while (lo < hi) {
                const int mid = (lo + hi) >> 1;
                if (cdf[mid] >= u) hi = mid; else lo = mid + 1;
            }
            atomicAdd(&cnt[lo], 1u);      // ds_add — row is block-private
        }
    }
    __syncthreads();

    float* orow = out2 + (long long)b * N_VIS;
    for (int x = tid; x < N_VIS; x += 256) orow[x] = (float)cnt[x];
}

// ---------------- host side -------------------------------------------------
extern "C" void kernel_launch(void* const* d_in, const int* in_sizes, int n_in,
                              void* d_out, int out_size, void* d_ws, size_t ws_size,
                              hipStream_t stream) {
    const int*   v      = (const int*)d_in[0];    // [512,15000] int32
    const float* W      = (const float*)d_in[1];  // [200,15000]
    const float* v_bias = (const float*)d_in[2];  // [15000]
    const float* h_bias = (const float*)d_in[3];  // [200]
    float* out  = (float*)d_out;
    float* out1 = out;                                  // float(v)
    float* out2 = out + (long long)BATCH * N_VIS;       // sampled v_

    // workspace layout (16B-aligned sections)
    char* p = (char*)d_ws;
    int*    dvec   = (int*)p;            p += 4096;
    float*  rowmax = (float*)p;          p += 4096;
    __bf16* hb     = (__bf16*)p;         p += (size_t)BATCH * KH * 2;   // 229 KB
    __bf16* vb     = (__bf16*)p;         p += (size_t)BATCH * KV * 2;   // 15.4 MB
    __bf16* wb     = (__bf16*)p;         p += (size_t)NHP * KV * 2;     // 6.2 MB
    __bf16* wt     = (__bf16*)p;         p += (size_t)KV * KH * 2;      // 6.7 MB
    float*  logits = (float*)p;                                         // 30.7 MB

    const long long nv = (long long)BATCH * N_VIS;

    k_rowsum<<<BATCH, 256, 0, stream>>>(v, dvec);
    k_cast<<<(unsigned)((nv + 255) / 256), 256, 0, stream>>>(v, out1, nv);
    k_prep_v <<<dim3((KV + 255) / 256, BATCH), 256, 0, stream>>>(v, vb);
    k_prep_wb<<<dim3((KV + 255) / 256, NHP),   256, 0, stream>>>(W, wb);
    k_prep_wt<<<dim3(KV / 16, KH / 16), 256, 0, stream>>>(W, wt);
    k_zero_hb<<<(BATCH * KH) / 256, 256, 0, stream>>>(hb);
    k_hidden<<<dim3(BATCH / 16, NHP / 16), 32, 0, stream>>>(vb, wb, h_bias, dvec, hb);
    k_logits<<<dim3(BATCH / 16, KV / 16), 32, 0, stream>>>(hb, wt, v_bias, logits);
    k_rowmax<<<BATCH, 256, 0, stream>>>(logits, rowmax);
    k_multinomial<<<BATCH, 256, (size_t)N_VIS * 8, stream>>>(logits, rowmax, dvec, out2);
}
